// Net_25864293057293
// MI455X (gfx1250) — compile-verified, matrix operand order checked by reading, not run on hardware
//
#include <hip/hip_runtime.h>

// ---------------------------------------------------------------------------
// 2-layer GCN forward for gfx1250 (MI455X).
//   h1 = relu( Agg( x @ W1 ) + b1 );  out = log_softmax( Agg( h1 @ W2 ) + b2 )
// GEMMs use v_wmma_f32_16x16x32_bf16 (bf16 A/B, f32 accum).
// Aggregation uses float4 gathers + global f32 atomics (bandwidth bound).
// ---------------------------------------------------------------------------

#define IN_FEATS 512
#define HIDDEN   64
#define NCLASSES 32

typedef __attribute__((ext_vector_type(16))) __bf16 v16bf;
typedef __attribute__((ext_vector_type(8)))  float  v8f;

union AFrag {
  v16bf v;
  unsigned short u[16];
};

__device__ __forceinline__ unsigned short f2b(float f) {
  // round-to-nearest-even float32 -> bf16
  unsigned int u = __float_as_uint(f);
  unsigned int r = u + 0x7fffu + ((u >> 16) & 1u);
  return (unsigned short)(r >> 16);
}

// ---------------------------------------------------------------- utilities
__global__ void fill_f32(float* __restrict__ p, long long n, float v) {
  long long i = blockIdx.x * (long long)blockDim.x + threadIdx.x;
  if (i < n) p[i] = v;
}

__global__ void degree_kernel(const int* __restrict__ ei, long long E,
                              float* __restrict__ deg) {
  long long e = blockIdx.x * (long long)blockDim.x + threadIdx.x;
  if (e < E) atomicAdd(&deg[ei[E + e]], 1.0f);
}

__global__ void rsqrt_inplace(float* __restrict__ p, long long n) {
  long long i = blockIdx.x * (long long)blockDim.x + threadIdx.x;
  if (i < n) {
    float d = p[i];
    p[i] = (d > 0.0f) ? rsqrtf(d) : 0.0f;
  }
}

// Pack W (K x NOUT, row-major f32) into bf16 WMMA-B fragment order.
// Fragment frag = nt*(K/32) + ks holds a 32x16 bf16 tile; per lane 16 values:
//   col = nt*16 + (lane&15)
//   lane<16 : K = ks*32 + {0..7, 16..23} ; lane>=16 : K = ks*32 + {8..15, 24..31}
__global__ void pack_w(const float* __restrict__ W, unsigned short* __restrict__ Wbf,
                       int K, int NOUT) {
  int idx = blockIdx.x * blockDim.x + threadIdx.x;
  int total = K * NOUT;
  if (idx >= total) return;
  int ksteps = K / 32;
  int j    = idx & 15;
  int lane = (idx >> 4) & 31;
  int frag = idx >> 9;
  int ks = frag % ksteps;
  int nt = frag / ksteps;
  int col = nt * 16 + (lane & 15);
  int kk = (j < 8) ? j : (j + 8);          // {0..7} or {16..23}
  int k = ks * 32 + ((lane < 16) ? kk : kk + 8);
  Wbf[idx] = f2b(W[k * NOUT + col]);
}

// ------------------------------------------------------------------- GEMM
// One wave computes a 16-row x NOUT output tile. A converted f32->bf16 on the
// fly; B read pre-packed (two b128 loads / fragment).
template <int K, int NOUT>
__global__ __launch_bounds__(256) void gemm_bf16_wmma(
    const float* __restrict__ X, const unsigned short* __restrict__ Wbf,
    float* __restrict__ H, int nrows) {
  constexpr int KSTEPS = K / 32;
  constexpr int NT = NOUT / 16;
  int wave = (int)((blockIdx.x * (long long)blockDim.x + threadIdx.x) >> 5);
  int lane = threadIdx.x & 31;
  int row0 = wave * 16;
  if (row0 >= nrows) return;
  int half = lane >> 4;     // 0: K {0..7,16..23}, 1: K {8..15,24..31}
  int rl = lane & 15;
  long long row = row0 + rl;

  v8f acc[NT];
#pragma unroll
  for (int nt = 0; nt < NT; ++nt) acc[nt] = (v8f){};

  for (int ks = 0; ks < KSTEPS; ++ks) {
    AFrag a;
    const float* xp = X + row * K + ks * 32 + half * 8;
    float4 f0 = *(const float4*)(xp + 0);
    float4 f1 = *(const float4*)(xp + 4);
    float4 f2 = *(const float4*)(xp + 16);
    float4 f3 = *(const float4*)(xp + 20);
    a.u[0] = f2b(f0.x);  a.u[1] = f2b(f0.y);  a.u[2] = f2b(f0.z);  a.u[3] = f2b(f0.w);
    a.u[4] = f2b(f1.x);  a.u[5] = f2b(f1.y);  a.u[6] = f2b(f1.z);  a.u[7] = f2b(f1.w);
    a.u[8] = f2b(f2.x);  a.u[9] = f2b(f2.y);  a.u[10] = f2b(f2.z); a.u[11] = f2b(f2.w);
    a.u[12] = f2b(f3.x); a.u[13] = f2b(f3.y); a.u[14] = f2b(f3.z); a.u[15] = f2b(f3.w);
#pragma unroll
    for (int nt = 0; nt < NT; ++nt) {
      int frag = nt * KSTEPS + ks;
      v16bf bv = *(const v16bf*)(Wbf + (long long)frag * 512 + lane * 16);
      acc[nt] = __builtin_amdgcn_wmma_f32_16x16x32_bf16(
          false, a.v, false, bv, (short)0, acc[nt], false, false);
    }
  }
  // C/D layout: VGPR m, lanes 0-15 -> row m, lanes 16-31 -> row m+8; col=lane&15
#pragma unroll
  for (int nt = 0; nt < NT; ++nt) {
#pragma unroll
    for (int m = 0; m < 8; ++m) {
      H[(long long)(row0 + m + half * 8) * NOUT + nt * 16 + rl] = acc[nt][m];
    }
  }
}

// -------------------------------------------------------------- scatter-add
// F/4 threads per edge; each handles one float4 feature chunk.
template <int F>
__global__ void scatter_kernel(const int* __restrict__ ei, long long E,
                               const float* __restrict__ dinv,
                               const float* __restrict__ Hin,
                               float* __restrict__ Agg) {
  constexpr int C = F / 4;
  long long t = blockIdx.x * (long long)blockDim.x + threadIdx.x;
  long long e = t / C;
  int c = (int)(t - e * C);
  if (e >= E) return;
  int s = ei[e];
  int d = ei[E + e];
  float nrm = dinv[s] * dinv[d];
  const float4 h = *(const float4*)(Hin + (long long)s * F + c * 4);
  float* o = Agg + (long long)d * F + c * 4;
  atomicAdd(o + 0, nrm * h.x);
  atomicAdd(o + 1, nrm * h.y);
  atomicAdd(o + 2, nrm * h.z);
  atomicAdd(o + 3, nrm * h.w);
}

// agg += dinv^2 * H (self loop) + bias, relu; in place (agg becomes h1)
__global__ void finalize_relu(float* __restrict__ agg, const float* __restrict__ H,
                              const float* __restrict__ dinv,
                              const float* __restrict__ bias, long long total) {
  long long i = blockIdx.x * (long long)blockDim.x + threadIdx.x;
  if (i >= total) return;
  long long n = i >> 6;       // / HIDDEN
  int f = (int)(i & (HIDDEN - 1));
  float di = dinv[n];
  float v = agg[i] + di * di * H[i] + bias[f];
  agg[i] = v > 0.0f ? v : 0.0f;
}

// One wave per node: 32 lanes == 32 classes. Self-loop + bias + log_softmax.
__global__ void finalize_logsoftmax(const float* __restrict__ agg,
                                    const float* __restrict__ H,
                                    const float* __restrict__ dinv,
                                    const float* __restrict__ bias,
                                    float* __restrict__ out, int N) {
  int node = (int)((blockIdx.x * (long long)blockDim.x + threadIdx.x) >> 5);
  int lane = threadIdx.x & 31;
  if (node >= N) return;
  float di = dinv[node];
  long long base = (long long)node * NCLASSES;
  float v = agg[base + lane] + di * di * H[base + lane] + bias[lane];
  float m = v;
#pragma unroll
  for (int o = 16; o > 0; o >>= 1) m = fmaxf(m, __shfl_xor(m, o, 32));
  float ex = expf(v - m);
  float s = ex;
#pragma unroll
  for (int o = 16; o > 0; o >>= 1) s += __shfl_xor(s, o, 32);
  out[base + lane] = (v - m) - logf(s);
}

// ---------------------------------------------------------------------------
static inline unsigned int nblk(long long n, int b) {
  return (unsigned int)((n + b - 1) / b);
}
static inline size_t align256(size_t x) { return (x + 255) & ~(size_t)255; }

extern "C" void kernel_launch(void* const* d_in, const int* in_sizes, int n_in,
                              void* d_out, int out_size, void* d_ws, size_t ws_size,
                              hipStream_t stream) {
  const float* X  = (const float*)d_in[0];
  const int*   EI = (const int*)d_in[1];
  const float* W1 = (const float*)d_in[2];
  const float* B1 = (const float*)d_in[3];
  const float* W2 = (const float*)d_in[4];
  const float* B2 = (const float*)d_in[5];
  float* OUT = (float*)d_out;

  const long long N = in_sizes[0] / IN_FEATS;   // 100000
  const long long E = in_sizes[1] / 2;          // 3200000

  // workspace layout
  char* ws = (char*)d_ws;
  size_t off = 0;
  float* dinv = (float*)(ws + off); off = align256(off + (size_t)N * 4);
  float* H1   = (float*)(ws + off); off = align256(off + (size_t)N * HIDDEN * 4);
  float* AGG1 = (float*)(ws + off); off = align256(off + (size_t)N * HIDDEN * 4);
  float* H2   = (float*)(ws + off); off = align256(off + (size_t)N * NCLASSES * 4);
  float* AGG2 = (float*)(ws + off); off = align256(off + (size_t)N * NCLASSES * 4);
  unsigned short* W1bf = (unsigned short*)(ws + off); off = align256(off + (size_t)IN_FEATS * HIDDEN * 2);
  unsigned short* W2bf = (unsigned short*)(ws + off); off = align256(off + (size_t)HIDDEN * NCLASSES * 2);
  (void)ws_size; (void)n_in; (void)out_size;

  const int B = 256;

  // init (must happen every call: scatter accumulates)
  fill_f32<<<nblk(N, B), B, 0, stream>>>(dinv, N, 1.0f);                 // deg incl. self loop
  fill_f32<<<nblk(N * HIDDEN, B), B, 0, stream>>>(AGG1, N * HIDDEN, 0.0f);
  fill_f32<<<nblk(N * NCLASSES, B), B, 0, stream>>>(AGG2, N * NCLASSES, 0.0f);

  // degree -> dinv
  degree_kernel<<<nblk(E, B), B, 0, stream>>>(EI, E, dinv);
  rsqrt_inplace<<<nblk(N, B), B, 0, stream>>>(dinv, N);

  // pre-pack weights into WMMA-B fragment order
  pack_w<<<nblk((long long)IN_FEATS * HIDDEN, B), B, 0, stream>>>(W1, W1bf, IN_FEATS, HIDDEN);
  pack_w<<<nblk((long long)HIDDEN * NCLASSES, B), B, 0, stream>>>(W2, W2bf, HIDDEN, NCLASSES);

  // layer 1: H1 = X @ W1  (WMMA bf16)
  {
    long long waves = (N + 15) / 16;
    gemm_bf16_wmma<IN_FEATS, HIDDEN><<<nblk(waves * 32, B), B, 0, stream>>>(X, W1bf, H1, (int)N);
  }
  // aggregate + relu
  scatter_kernel<HIDDEN><<<nblk(E * (HIDDEN / 4), B), B, 0, stream>>>(EI, E, dinv, H1, AGG1);
  finalize_relu<<<nblk(N * HIDDEN, B), B, 0, stream>>>(AGG1, H1, dinv, B1, N * HIDDEN);

  // layer 2: H2 = h1 @ W2  (WMMA bf16)
  {
    long long waves = (N + 15) / 16;
    gemm_bf16_wmma<HIDDEN, NCLASSES><<<nblk(waves * 32, B), B, 0, stream>>>(AGG1, W2bf, H2, (int)N);
  }
  scatter_kernel<NCLASSES><<<nblk(E * (NCLASSES / 4), B), B, 0, stream>>>(EI, E, dinv, H2, AGG2);

  // bias + self loop + log_softmax (32 lanes == 32 classes per wave)
  finalize_logsoftmax<<<nblk(N * 32, B), B, 0, stream>>>(AGG2, H2, dinv, B2, OUT, (int)N);
}
